// Top_down_90469191123500
// MI455X (gfx1250) — compile-verified, matrix operand order checked by reading, not run on hardware
//
#include <hip/hip_runtime.h>
#include <hip/hip_bf16.h>

typedef __attribute__((ext_vector_type(16))) __bf16 v16bf;
typedef __attribute__((ext_vector_type(8)))  float  v8f;

#define CCH 256            // channels
#define KDIM 2304          // C*9 reduction dim
#define CHUNK 16384        // max pixels per GEMM chunk

// ---------------------------------------------------------------------------
// 1) fuse: xnhwc[b][y][x][c] = x_nchw + nearest(t_prev)   (fp32, NHWC)
// ---------------------------------------------------------------------------
__global__ void fuse_kernel(const float* __restrict__ x,
                            const float* __restrict__ tprev,  // may be null
                            float* __restrict__ xnhwc,
                            int H, int W, int Hp, int Wp) {
    int p = blockIdx.x;            // pixel index over B*H*W
    int c = threadIdx.x;           // channel
    int HW = H * W;
    int b = p / HW;
    int rem = p - b * HW;
    int y = rem / W;
    int xx = rem - y * W;
    float v = x[((size_t)(b * CCH + c) * H + y) * W + xx];
    if (tprev) {
        int iy = (y * Hp) / H;
        int ix = (xx * Wp) / W;
        v += tprev[((size_t)(b * CCH + c) * Hp + iy) * Wp + ix];
    }
    xnhwc[(size_t)p * CCH + c] = v;
}

// ---------------------------------------------------------------------------
// 2a) prep main weights: OIHW fp32 -> A_bf16[o][tap*256 + c]
// ---------------------------------------------------------------------------
__global__ void prep_w_kernel(const float* __restrict__ w, __bf16* __restrict__ A) {
    int idx = blockIdx.x * blockDim.x + threadIdx.x;      // [0, 256*2304)
    if (idx >= CCH * KDIM) return;
    int o = idx / KDIM;
    int r = idx - o * KDIM;
    int tap = r / CCH;
    int c = r - tap * CCH;
    int ky = tap / 3, kx = tap - ky * 3;
    A[idx] = (__bf16)w[(((size_t)o * CCH + c) * 3 + ky) * 3 + kx];
}

// 2b) prep offset weights: [18][C][3][3] fp32 -> offwt[(j*9+tap)*256 + c] fp32
__global__ void prep_offw_kernel(const float* __restrict__ w, float* __restrict__ T) {
    int idx = blockIdx.x * blockDim.x + threadIdx.x;      // [0, 18*2304)
    if (idx >= 18 * KDIM) return;
    int j = idx / KDIM;
    int r = idx - j * KDIM;
    int tap = r / CCH;
    int c = r - tap * CCH;
    int ky = tap / 3, kx = tap - ky * 3;
    T[idx] = w[(((size_t)j * CCH + c) * 3 + ky) * 3 + kx];
}

// ---------------------------------------------------------------------------
// 3) offset conv: 3x3, C=256 -> 18, zero pad. Weights staged in 166KB LDS.
//    offsets_out[b][y][x][ch 0..17]
// ---------------------------------------------------------------------------
__global__ void offset_conv_kernel(const float* __restrict__ xnhwc,
                                   const float* __restrict__ offwt, // [18*2304]
                                   const float* __restrict__ offb,  // [18]
                                   float* __restrict__ offsets,
                                   int H, int W, int Npix) {
    extern __shared__ float smem[];                       // 18*2304 floats
    for (int i = threadIdx.x; i < 18 * KDIM; i += blockDim.x)
        smem[i] = offwt[i];
    __syncthreads();

    int p = blockIdx.x * blockDim.x + threadIdx.x;
    if (p >= Npix) return;
    int HW = H * W;
    int b = p / HW;
    int rem = p - b * HW;
    int y = rem / W;
    int xx = rem - y * W;

    float acc[18];
#pragma unroll
    for (int j = 0; j < 18; ++j) acc[j] = offb[j];

    for (int tap = 0; tap < 9; ++tap) {
        int yy = y + tap / 3 - 1;
        int xw = xx + tap % 3 - 1;
        if (yy < 0 || yy >= H || xw < 0 || xw >= W) continue;
        const float* src = xnhwc + ((size_t)(b * H + yy) * W + xw) * CCH;
        for (int c = 0; c < CCH; ++c) {
            float v = src[c];
#pragma unroll
            for (int j = 0; j < 18; ++j)
                acc[j] += v * smem[(j * 9 + tap) * CCH + c];
        }
    }
    float* dst = offsets + (size_t)p * 18;
#pragma unroll
    for (int j = 0; j < 18; ++j) dst[j] = acc[j];
}

// ---------------------------------------------------------------------------
// 4) im2col (deformable bilinear, or plain zero-pad shift if offsets==null)
//    Bbuf[n_local][tap*256 + c] bf16
// ---------------------------------------------------------------------------
__global__ void im2col_kernel(const float* __restrict__ xnhwc,
                              const float* __restrict__ offsets, // null => plain conv
                              __bf16* __restrict__ Bbuf,
                              int H, int W, int n0) {
    int n_local = blockIdx.x;
    int n = n0 + n_local;
    int c = threadIdx.x;
    int HW = H * W;
    int b = n / HW;
    int rem = n - b * HW;
    int y = rem / W;
    int xx = rem - y * W;
    const float* xb = xnhwc + (size_t)b * HW * CCH;
    __bf16* dst = Bbuf + (size_t)n_local * KDIM;

    for (int tap = 0; tap < 9; ++tap) {
        float dy = 0.f, dx = 0.f;
        if (offsets) {
            const float* ob = offsets + (size_t)(rem + b * HW) * 18;
            dy = ob[2 * tap];
            dx = ob[2 * tap + 1];
        }
        float py = (float)y + (float)(tap / 3 - 1) + dy;
        float px = (float)xx + (float)(tap % 3 - 1) + dx;
        float y0f = floorf(py), x0f = floorf(px);
        int y0 = (int)y0f, x0 = (int)x0f;
        float wy1 = py - y0f, wx1 = px - x0f;
        float wy0 = 1.f - wy1, wx0 = 1.f - wx1;

        float acc = 0.f;
#pragma unroll
        for (int q = 0; q < 4; ++q) {
            int yi = y0 + (q >> 1);
            int xi = x0 + (q & 1);
            float wq = ((q >> 1) ? wy1 : wy0) * ((q & 1) ? wx1 : wx0);
            if (yi >= 0 && yi < H && xi >= 0 && xi < W)
                acc += wq * xb[((size_t)yi * W + xi) * CCH + c];
        }
        dst[tap * CCH + c] = (__bf16)acc;
    }
}

// ---------------------------------------------------------------------------
// 5) WMMA GEMM: out[o][n] = sum_k A[o][k]*B[n][k] + bias[o]
//    Each wave: one 16-wide N tile, all 16 M tiles (M=256). bf16 -> f32 acc.
// ---------------------------------------------------------------------------
__global__ __launch_bounds__(256) void gemm_kernel(
        const __bf16* __restrict__ A,    // [256][2304]
        const __bf16* __restrict__ Bm,   // [chunkN][2304]
        const float*  __restrict__ bias, // [256]
        float* __restrict__ out,         // level output base, NCHW fp32
        int chunkN, int n0, int HW) {
    int wave = threadIdx.x >> 5;
    int lane = threadIdx.x & 31;
    int ntile = blockIdx.x * 8 + wave;
    int n_base = ntile * 16;
    if (n_base >= chunkN) return;      // wave-uniform: EXEC stays all-ones

    int row = lane & 15;               // fragment row (M for A, N for B)
    int hi  = lane >> 4;               // K half: 0 -> K0..15, 1 -> K16..31

    const __bf16* bptr = Bm + (size_t)(n_base + row) * KDIM + hi * 16;
    const __bf16* abase = A + (size_t)row * KDIM + hi * 16;

    v8f acc[16];
    v8f zero = {};
#pragma unroll
    for (int t = 0; t < 16; ++t) acc[t] = zero;

    for (int k = 0; k < KDIM; k += 32) {
        v16bf bfrag = *(const v16bf*)(bptr + k);
#pragma unroll
        for (int t = 0; t < 16; ++t) {
            v16bf afrag = *(const v16bf*)(abase + (size_t)t * 16 * KDIM + k);
            acc[t] = __builtin_amdgcn_wmma_f32_16x16x32_bf16(
                false, afrag, false, bfrag, (short)0, acc[t], false, false);
        }
    }

    // Epilogue: D layout -> lane: N = lane&15, rows M = r + 8*(lane>>4)
    int col = lane & 15;
    int rowoff = (lane >> 4) * 8;
    int n = n0 + n_base + col;
    int b = n / HW;
    int p = n - b * HW;
    float* obase = out + (size_t)b * CCH * HW + p;
#pragma unroll
    for (int t = 0; t < 16; ++t) {
#pragma unroll
        for (int r = 0; r < 8; ++r) {
            int o = t * 16 + rowoff + r;
            obase[(size_t)o * HW] = acc[t][r] + bias[o];
        }
    }
}

// ---------------------------------------------------------------------------
// Host driver
// ---------------------------------------------------------------------------
extern "C" void kernel_launch(void* const* d_in, const int* in_sizes, int n_in,
                              void* d_out, int out_size, void* d_ws, size_t ws_size,
                              hipStream_t stream) {
    const float* xin[4] = {(const float*)d_in[0], (const float*)d_in[1],
                           (const float*)d_in[2], (const float*)d_in[3]};
    const float* weights = (const float*)d_in[4];  // [4][256][256][3][3]
    const float* biases  = (const float*)d_in[5];  // [4][256]
    const float* off_w   = (const float*)d_in[6];  // [4][18][256][3][3]
    const float* off_b   = (const float*)d_in[7];  // [4][18]
    float* out = (float*)d_out;

    const int Hs[4] = {128, 64, 32, 16};
    const size_t out_off[4] = {0, 33554432, 41943040, 44040192};

    // workspace carve (256B aligned)
    char* ws = (char*)d_ws;
    size_t o = 0;
    auto carve = [&](size_t bytes) { char* p = ws + o; o = (o + bytes + 255) & ~(size_t)255; return p; };
    float*  xnhwc   = (float*)carve((size_t)8 * 128 * 128 * CCH * 4);  // 128 MB
    float*  offsets = (float*)carve((size_t)8 * 64 * 64 * 18 * 4);     // max deform level (L1)
    __bf16* Abf     = (__bf16*)carve((size_t)CCH * KDIM * 2);
    float*  offwt   = (float*)carve((size_t)18 * KDIM * 4);
    __bf16* Bbuf    = (__bf16*)carve((size_t)CHUNK * KDIM * 2);        // 72 MB

    for (int lvl = 3; lvl >= 0; --lvl) {
        int H = Hs[lvl], W = Hs[lvl];
        int HW = H * W;
        int N = 8 * HW;
        const float* tprev = (lvl == 3) ? nullptr : out + out_off[lvl + 1];
        int Hp = (lvl == 3) ? 1 : Hs[lvl + 1];

        fuse_kernel<<<N, CCH, 0, stream>>>(xin[lvl], tprev, xnhwc, H, W, Hp, Hp);

        if (lvl > 0) {
            prep_offw_kernel<<<(18 * KDIM + 255) / 256, 256, 0, stream>>>(
                off_w + (size_t)lvl * 18 * KDIM, offwt);
            offset_conv_kernel<<<(N + 255) / 256, 256, 18 * KDIM * 4, stream>>>(
                xnhwc, offwt, off_b + lvl * 18, offsets, H, W, N);
        }

        prep_w_kernel<<<(CCH * KDIM + 255) / 256, 256, 0, stream>>>(
            weights + (size_t)lvl * CCH * KDIM, Abf);

        for (int n0 = 0; n0 < N; n0 += CHUNK) {
            int cN = (N - n0 < CHUNK) ? (N - n0) : CHUNK;
            im2col_kernel<<<cN, CCH, 0, stream>>>(
                xnhwc, (lvl > 0) ? offsets : nullptr, Bbuf, H, W, n0);
            int ntiles = cN / 16;
            gemm_kernel<<<(ntiles + 7) / 8, 256, 0, stream>>>(
                Abf, Bbuf, biases + lvl * CCH, out + out_off[lvl], cN, n0, HW);
        }
    }
}
